// UMambaConvBlockAnt_32916629357148
// MI455X (gfx1250) — matrix operand order repaired and checked by reading, not compiled, not run on hardware
//
#include <hip/hip_runtime.h>
#include <cmath>

typedef __bf16 bf16_t;
typedef __attribute__((ext_vector_type(8)))  bf16_t v8bf;
typedef __attribute__((ext_vector_type(16))) bf16_t v16bf;
typedef __attribute__((ext_vector_type(8)))  float  v8f;
typedef __attribute__((ext_vector_type(4)))  unsigned int v4u;
typedef __attribute__((ext_vector_type(8)))  int v8i;
typedef __attribute__((ext_vector_type(4)))  int v4i;

#define NTOK 32768   // B*L = 8*4096
#define LSEQ 4096

static __device__ __forceinline__ float silu_f(float v){
  return v * __builtin_amdgcn_rcpf(1.f + __expf(-v));
}

static __device__ __forceinline__ v8f v8f_zero(){
  v8f z = {0.f,0.f,0.f,0.f,0.f,0.f,0.f,0.f};
  return z;
}

static __device__ __forceinline__ v8bf pack8(const float* v){
  v8bf p;
#pragma unroll
  for (int i = 0; i < 8; ++i) p[i] = (bf16_t)v[i];
  return p;
}

static __device__ __forceinline__ v8f wmma_bf16(v16bf a, v16bf b, v8f c){
  // D = A(16x32 bf16) x B(32x16 bf16) + C(f32)
  return __builtin_amdgcn_wmma_f32_16x16x32_bf16(false, a, false, b, (short)0, c, false, false);
}

// A fragment: lane l -> M = m0 + (l&15); elements e<8: K = 8h+e ; e>=8: K = 16+8h+(e-8)
static __device__ __forceinline__ v16bf frag_a(const bf16_t* As, int m0, int lane){
  int row = m0 + (lane & 15), h = lane >> 4;
  const bf16_t* base = As + row * 32 + 8 * h;
  v8bf lo = *(const v8bf*)(base);
  v8bf hi = *(const v8bf*)(base + 16);
  v16bf r;
#pragma unroll
  for (int i = 0; i < 8; ++i){ r[i] = lo[i]; r[i + 8] = hi[i]; }
  return r;
}
// B fragment: lane l -> N = n0 + (l&15); elements e<8: K = 8h+e ; e>=8: K = 16+8h+(e-8)   (Ws stored [n][k], ld=32)
static __device__ __forceinline__ v16bf frag_b(const bf16_t* Ws, int n0, int lane){
  int col = n0 + (lane & 15), h = lane >> 4;
  const bf16_t* base = Ws + col * 32 + 8 * h;
  v8bf lo = *(const v8bf*)(base);
  v8bf hi = *(const v8bf*)(base + 16);
  v16bf r;
#pragma unroll
  for (int i = 0; i < 8; ++i){ r[i] = lo[i]; r[i + 8] = hi[i]; }
  return r;
}

// ---- staging: each thread owns 8 consecutive K -> one ds_store_b128 per thread ----

// As[t][k] (64x32 bf16) <- transpose from channel-major f32 src[(k0+k)*ld + c0 + t]
static __device__ __forceinline__ void stage_at(bf16_t* As, const float* __restrict__ src,
                                                long ld, int k0, long c0, int tid){
  int t = tid & 63, kb = (tid >> 6) << 3;          // k = kb + i
  const float* p = src + (long)(k0 + kb) * ld + c0 + t;
  float v[8];
#pragma unroll
  for (int i = 0; i < 8; ++i) v[i] = p[(long)i * ld];
  *(v8bf*)(As + t * 32 + kb) = pack8(v);
}
// K valid only 0..7 (dt_proj): only the kb==0 quarter loads, rest zero-fill
static __device__ __forceinline__ void stage_at_gk8(bf16_t* As, const float* __restrict__ src,
                                                    long ld, long c0, int tid){
  int t = tid & 63, kb = (tid >> 6) << 3;
  float v[8];
  if (kb == 0){
#pragma unroll
    for (int i = 0; i < 8; ++i) v[i] = src[(long)i * ld + c0 + t];
  } else {
#pragma unroll
    for (int i = 0; i < 8; ++i) v[i] = 0.f;
  }
  *(v8bf*)(As + t * 32 + kb) = pack8(v);
}
// Ws[n][k] <- W[(n0+n)*ldw + k0+k]; RITER*64 rows, fully valid; vector global loads
template<int RITER>
static __device__ __forceinline__ void stage_w_full(bf16_t* Ws, const float* __restrict__ W,
                                                    int ldw, int n0, int k0, int tid){
  int kb = (tid & 3) << 3, nr = tid >> 2;
#pragma unroll
  for (int j = 0; j < RITER; ++j){
    int n = nr + j * 64;
    const float* p = W + (long)(n0 + n) * ldw + k0 + kb;
    float v[8];
#pragma unroll
    for (int i = 0; i < 8; ++i) v[i] = p[i];
    *(v8bf*)(Ws + n * 32 + kb) = pack8(v);
  }
}
// N-guarded (x_proj: nvalid=136) -- guard is uniform per thread
static __device__ __forceinline__ void stage_w_gn(bf16_t* Ws, const float* __restrict__ W,
                                                  int ldw, int n0, int k0, int nvalid, int tid){
  int kb = (tid & 3) << 3, n = tid >> 2;
  float v[8];
  if (n0 + n < nvalid){
    const float* p = W + (long)(n0 + n) * ldw + k0 + kb;
#pragma unroll
    for (int i = 0; i < 8; ++i) v[i] = p[i];
  } else {
#pragma unroll
    for (int i = 0; i < 8; ++i) v[i] = 0.f;
  }
  *(v8bf*)(Ws + n * 32 + kb) = pack8(v);
}
// K valid 0..7 == full row (dt_proj: ldw=8)
static __device__ __forceinline__ void stage_w_gk8(bf16_t* Ws, const float* __restrict__ W,
                                                   int ldw, int n0, int tid){
  int kb = (tid & 3) << 3, n = tid >> 2;
  float v[8];
  if (kb == 0){
    const float* p = W + (long)(n0 + n) * ldw;
#pragma unroll
    for (int i = 0; i < 8; ++i) v[i] = p[i];
  } else {
#pragma unroll
    for (int i = 0; i < 8; ++i) v[i] = 0.f;
  }
  *(v8bf*)(Ws + n * 32 + kb) = pack8(v);
}

// ---------------- Kernel 1: t1 = silu(x_tok @ proj_in_w^T + b); tokens x 128, bf16 token-major
__global__ void k_proj_in(const float* __restrict__ x, const float* __restrict__ W1,
                          const float* __restrict__ b1, bf16_t* __restrict__ t1){
  __shared__ __align__(16) bf16_t As[2048];
  __shared__ __align__(16) bf16_t Ws[2048];
  int tid = threadIdx.x, lane = tid & 31, w = tid >> 5;
  long t0 = (long)blockIdx.x * 64;
  int b  = (int)(t0 >> 12);
  long l0 = t0 & 4095;
  int gn0 = blockIdx.y * 64;
  int mt = w >> 1, nb = (w & 1) * 2;
  v8f acc[2] = {v8f_zero(), v8f_zero()};
  for (int kc = 0; kc < 2; ++kc){
    __syncthreads();
    stage_at(As, x + (long)b * 64 * 4096, 4096, kc * 32, l0, tid);
    stage_w_full<1>(Ws, W1, 64, gn0, kc * 32, tid);
    __syncthreads();
    v16bf af = frag_a(As, mt * 16, lane);
#pragma unroll
    for (int i = 0; i < 2; ++i)
      acc[i] = wmma_bf16(af, frag_b(Ws, (nb + i) * 16, lane), acc[i]);
  }
  int h = lane >> 4, cl = lane & 15;
#pragma unroll
  for (int i = 0; i < 2; ++i){
    int col = gn0 + (nb + i) * 16 + cl;
    float bias = b1[col];
    long trow = t0 + mt * 16 + 8 * h;
#pragma unroll
    for (int r = 0; r < 8; ++r)
      t1[(trow + r) * 128 + col] = (bf16_t)silu_f(acc[i][r] + bias);
  }
}

// ---------------- Kernel 2: xz = t1 @ in_proj_w^T ; split -> xmrT [256][NTOK], zT [256][NTOK]
// A-tile (64x32 bf16 tile of a 32768x128 bf16 tensor) staged via the Tensor Data Mover.
__global__ void k_in_proj(const bf16_t* __restrict__ t1, const float* __restrict__ W2,
                          float* __restrict__ xmrT, float* __restrict__ zT){
  __shared__ __align__(16) bf16_t As[2048];
  __shared__ __align__(16) bf16_t Ws[2048];
  int tid = threadIdx.x, lane = tid & 31, w = tid >> 5;
  long t0 = (long)blockIdx.x * 64;
  int gn0 = blockIdx.y * 64;
  int mt = w >> 1, nb = (w & 1) * 2;
  v8f acc[2] = {v8f_zero(), v8f_zero()};
  for (int kc = 0; kc < 4; ++kc){
    __syncthreads();
#if __has_builtin(__builtin_amdgcn_tensor_load_to_lds) && __has_builtin(__builtin_amdgcn_s_wait_tensorcnt)
    if (tid < 32){
      // D# per CDNA5 ISA 8.3/8.4: 2D tensor 32768 x 128 bf16, tile 64 rows x 32 cols
      unsigned ldsa = (unsigned)(uintptr_t)(void*)As;
      unsigned long long ga =
          (unsigned long long)(uintptr_t)(const void*)(t1 + t0 * 128 + kc * 32);
      v4u g0;
      g0[0] = 1u;                                           // count=1 (valid user D#)
      g0[1] = ldsa;                                         // lds_addr (bytes)
      g0[2] = (unsigned)(ga & 0xFFFFFFFFu);                 // global_addr[31:0]
      g0[3] = (unsigned)((ga >> 32) & 0x01FFFFFFu) | 0x80000000u; // addr[56:32] | type=2
      v8i g1;
      g1[0] = 0x00010000;                                   // data_size=1 (2 bytes)
      g1[1] = (int)((128u & 0xFFFFu) << 16);                // tensor_dim0=128 (low 16)
      g1[2] = (int)(((unsigned)NTOK & 0xFFFFu) << 16);      // tensor_dim1=32768 (low 16)
      g1[3] = (int)(32u << 16);                             // tile_dim0=32
      g1[4] = 64;                                           // tile_dim1=64
      g1[5] = 128;                                          // tensor_dim0_stride=128
      g1[6] = 0;
      g1[7] = 0;
      v4i zz4 = {0, 0, 0, 0};
      v8i zz8 = {0, 0, 0, 0, 0, 0, 0, 0};
      __builtin_amdgcn_tensor_load_to_lds(g0, g1, zz4, zz4, zz8, 0);
      __builtin_amdgcn_s_wait_tensorcnt(0);
    }
#else
    {
      int row = tid >> 2, chunk = tid & 3;                  // 16B per thread
      *(uint4*)(As + row * 32 + chunk * 8) =
          *(const uint4*)(t1 + (t0 + row) * 128 + kc * 32 + chunk * 8);
    }
#endif
    stage_w_full<1>(Ws, W2, 128, gn0, kc * 32, tid);
    __syncthreads();
    v16bf af = frag_a(As, mt * 16, lane);
#pragma unroll
    for (int i = 0; i < 2; ++i)
      acc[i] = wmma_bf16(af, frag_b(Ws, (nb + i) * 16, lane), acc[i]);
  }
  int h = lane >> 4, cl = lane & 15;
#pragma unroll
  for (int i = 0; i < 2; ++i){
    int col = gn0 + (nb + i) * 16 + cl;
    long trow = t0 + mt * 16 + 8 * h;
    float* dst = (col < 256) ? (xmrT + (long)col * NTOK) : (zT + (long)(col - 256) * NTOK);
#pragma unroll
    for (int r = 0; r < 8; ++r) dst[trow + r] = acc[i][r];
  }
}

// ---------------- Kernel 3: depthwise causal conv1d (D_CONV=4) + silu -> xmaT [256][NTOK]
__global__ void k_conv1d(const float* __restrict__ xmrT, const float* __restrict__ cw,
                         const float* __restrict__ cb, float* __restrict__ xmaT){
  __shared__ float row[4096];
  int d = blockIdx.x >> 3, b = blockIdx.x & 7;
  const float* src = xmrT + (long)d * NTOK + b * 4096;
  for (int i = threadIdx.x; i < 4096; i += 256) row[i] = src[i];
  __syncthreads();
  float w0 = cw[d * 4 + 0], w1 = cw[d * 4 + 1], w2 = cw[d * 4 + 2], w3 = cw[d * 4 + 3];
  float bb = cb[d];
  float* dst = xmaT + (long)d * NTOK + b * 4096;
  for (int l = threadIdx.x; l < 4096; l += 256){
    float acc = bb + w3 * row[l];
    if (l >= 1) acc += w2 * row[l - 1];
    if (l >= 2) acc += w1 * row[l - 2];
    if (l >= 3) acc += w0 * row[l - 3];
    dst[l] = silu_f(acc);
  }
}

// ---------------- Kernel 4: dbl = xm @ x_proj_w^T (N=136) ; split dt-low / B / C
__global__ void k_x_proj(const float* __restrict__ xmaT, const float* __restrict__ Wx,
                         float* __restrict__ dtlowT, float* __restrict__ Bc, float* __restrict__ Cc){
  __shared__ __align__(16) bf16_t As[2048];
  __shared__ __align__(16) bf16_t Ws[2048];
  int tid = threadIdx.x, lane = tid & 31, w = tid >> 5;
  long t0 = (long)blockIdx.x * 64;
  int gn0 = blockIdx.y * 64;
  int mt = w >> 1, nb = (w & 1) * 2;
  v8f acc[2] = {v8f_zero(), v8f_zero()};
  for (int kc = 0; kc < 8; ++kc){
    __syncthreads();
    stage_at(As, xmaT, NTOK, kc * 32, t0, tid);
    stage_w_gn(Ws, Wx, 256, gn0, kc * 32, 136, tid);
    __syncthreads();
    v16bf af = frag_a(As, mt * 16, lane);
#pragma unroll
    for (int i = 0; i < 2; ++i)
      acc[i] = wmma_bf16(af, frag_b(Ws, (nb + i) * 16, lane), acc[i]);
  }
  int h = lane >> 4, cl = lane & 15;
#pragma unroll
  for (int i = 0; i < 2; ++i){
    int col = gn0 + (nb + i) * 16 + cl;
    if (col >= 136) continue;
    long trow = t0 + mt * 16 + 8 * h;
    if (col < 8){
      for (int r = 0; r < 8; ++r) dtlowT[(long)col * NTOK + trow + r] = acc[i][r];
    } else if (col < 72){
      int n = col - 8;
      for (int r = 0; r < 8; ++r) Bc[(trow + r) * 64 + n] = acc[i][r];
    } else {
      int n = col - 72;
      for (int r = 0; r < 8; ++r) Cc[(trow + r) * 64 + n] = acc[i][r];
    }
  }
}

// ---------------- Kernel 5: dt = softplus(dtlow @ dt_proj_w^T + b) -> dtT [256][NTOK]
__global__ void k_dt_proj(const float* __restrict__ dtlowT, const float* __restrict__ Wdt,
                          const float* __restrict__ bdt, float* __restrict__ dtT){
  __shared__ __align__(16) bf16_t As[2048];
  __shared__ __align__(16) bf16_t Ws[2048];
  int tid = threadIdx.x, lane = tid & 31, w = tid >> 5;
  long t0 = (long)blockIdx.x * 64;
  int gn0 = blockIdx.y * 64;
  int mt = w >> 1, nb = (w & 1) * 2;
  v8f acc[2] = {v8f_zero(), v8f_zero()};
  stage_at_gk8(As, dtlowT, NTOK, t0, tid);            // K padded 8 -> 32 with zeros
  stage_w_gk8(Ws, Wdt, 8, gn0, tid);
  __syncthreads();
  {
    v16bf af = frag_a(As, mt * 16, lane);
#pragma unroll
    for (int i = 0; i < 2; ++i)
      acc[i] = wmma_bf16(af, frag_b(Ws, (nb + i) * 16, lane), acc[i]);
  }
  int h = lane >> 4, cl = lane & 15;
#pragma unroll
  for (int i = 0; i < 2; ++i){
    int col = gn0 + (nb + i) * 16 + cl;
    float bias = bdt[col];
    long trow = t0 + mt * 16 + 8 * h;
#pragma unroll
    for (int r = 0; r < 8; ++r){
      float v = acc[i][r] + bias;
      v = (v > 20.f) ? v : log1pf(__expf(v));         // softplus
      dtT[(long)col * NTOK + trow + r] = v;
    }
  }
}

// ---------------- Kernel 6: selective scan. One wave per (d,b); 64 states / 32 lanes.
__global__ void k_scan(const float* __restrict__ dtT, const float* __restrict__ xmaT,
                       const float* __restrict__ Bc, const float* __restrict__ Cc,
                       const float* __restrict__ A_log, float* __restrict__ yT){
  int wv = (blockIdx.x << 3) | (threadIdx.x >> 5);    // 0..2047
  int lane = threadIdx.x & 31;
  int d = wv >> 3, b = wv & 7;
  float a0 = -__expf(A_log[d * 64 + lane]);
  float a1 = -__expf(A_log[d * 64 + lane + 32]);
  long base = (long)d * NTOK + b * 4096;
  const float* dtp = dtT + base;
  const float* up  = xmaT + base;
  const float* bp  = Bc + (long)b * 4096 * 64;
  const float* cp  = Cc + (long)b * 4096 * 64;
  float* yo = yT + base;
  float h0 = 0.f, h1 = 0.f;
  for (int l = 0; l < 4096; ++l){
    float dtv = dtp[l], u = up[l];
    float bc0 = bp[l * 64 + lane], bc1 = bp[l * 64 + lane + 32];
    float cc0 = cp[l * 64 + lane], cc1 = cp[l * 64 + lane + 32];
    float du = dtv * u;
    h0 = h0 * __expf(dtv * a0) + du * bc0;
    h1 = h1 * __expf(dtv * a1) + du * bc1;
    float part = h0 * cc0 + h1 * cc1;
#pragma unroll
    for (int off = 16; off > 0; off >>= 1) part += __shfl_xor(part, off, 32);
    if (lane == 0) yo[l] = part;
  }
}

// ---------------- Kernel 7: gate = (y + xm*D) * silu(z) ; o1 = gate @ out_proj_w^T -> o1T [128][NTOK]
__global__ void k_out_proj(const float* __restrict__ yT, const float* __restrict__ xmaT,
                           const float* __restrict__ zT, const float* __restrict__ Dp,
                           const float* __restrict__ Wo, float* __restrict__ o1T){
  __shared__ __align__(16) bf16_t As[2048];
  __shared__ __align__(16) bf16_t Ws[2048];
  int tid = threadIdx.x, lane = tid & 31, w = tid >> 5;
  long t0 = (long)blockIdx.x * 64;
  int gn0 = blockIdx.y * 64;
  int mt = w >> 1, nb = (w & 1) * 2;
  v8f acc[2] = {v8f_zero(), v8f_zero()};
  for (int kc = 0; kc < 8; ++kc){
    __syncthreads();
    {  // fused gating stage, two-phase, one ds_store_b128 per thread
      int t = tid & 63, kb = (tid >> 6) << 3;
      long tok = t0 + t;
      long dch = (long)(kc * 32 + kb);
      const float* yp = yT   + dch * NTOK + tok;
      const float* xp = xmaT + dch * NTOK + tok;
      const float* zp = zT   + dch * NTOK + tok;
      float yv[8], xv[8], zv[8], dv[8];
#pragma unroll
      for (int i = 0; i < 8; ++i){
        yv[i] = yp[(long)i * NTOK];
        xv[i] = xp[(long)i * NTOK];
        zv[i] = zp[(long)i * NTOK];
        dv[i] = Dp[dch + i];
      }
      float gv[8];
#pragma unroll
      for (int i = 0; i < 8; ++i)
        gv[i] = (yv[i] + xv[i] * dv[i]) * silu_f(zv[i]);
      *(v8bf*)(As + t * 32 + kb) = pack8(gv);
    }
    stage_w_full<1>(Ws, Wo, 256, gn0, kc * 32, tid);
    __syncthreads();
    v16bf af = frag_a(As, mt * 16, lane);
#pragma unroll
    for (int i = 0; i < 2; ++i)
      acc[i] = wmma_bf16(af, frag_b(Ws, (nb + i) * 16, lane), acc[i]);
  }
  int h = lane >> 4, cl = lane & 15;
#pragma unroll
  for (int i = 0; i < 2; ++i){
    int col = gn0 + (nb + i) * 16 + cl;
    long trow = t0 + mt * 16 + 8 * h;
#pragma unroll
    for (int r = 0; r < 8; ++r) o1T[(long)col * NTOK + trow + r] = acc[i][r];
  }
}

// ---------------- Kernel 8: t = o1 @ proj_out_w^T + b ; LayerNorm(128) -> fT [128][NTOK]
__global__ void k_proj_out_ln(const float* __restrict__ o1T, const float* __restrict__ Wp,
                              const float* __restrict__ bp, const float* __restrict__ ln_g,
                              const float* __restrict__ ln_b, float* __restrict__ fT){
  __shared__ __align__(16) bf16_t As[2048];
  __shared__ __align__(16) bf16_t Ws[4096];
  __shared__ float Cls[64 * 129];
  __shared__ float redS[256], redQ[256];
  __shared__ float muS[64], rsS[64];
  int tid = threadIdx.x, lane = tid & 31, w = tid >> 5;
  long t0 = (long)blockIdx.x * 64;
  int mt = w >> 1, nb = (w & 1) * 4;
  v8f acc[4] = {v8f_zero(), v8f_zero(), v8f_zero(), v8f_zero()};
  for (int kc = 0; kc < 4; ++kc){
    __syncthreads();
    stage_at(As, o1T, NTOK, kc * 32, t0, tid);
    stage_w_full<2>(Ws, Wp, 128, 0, kc * 32, tid);
    __syncthreads();
    v16bf af = frag_a(As, mt * 16, lane);
#pragma unroll
    for (int i = 0; i < 4; ++i)
      acc[i] = wmma_bf16(af, frag_b(Ws, (nb + i) * 16, lane), acc[i]);
  }
  int h = lane >> 4, cl = lane & 15;
#pragma unroll
  for (int i = 0; i < 4; ++i){
    int col = (nb + i) * 16 + cl;
    float bias = bp[col];
#pragma unroll
    for (int r = 0; r < 8; ++r)
      Cls[(mt * 16 + 8 * h + r) * 129 + col] = acc[i][r] + bias;
  }
  __syncthreads();
  int tok = tid >> 2, q = tid & 3;
  float s = 0.f, sq = 0.f;
  for (int c = q * 32; c < q * 32 + 32; ++c){
    float v = Cls[tok * 129 + c];
    s += v; sq += v * v;
  }
  redS[tok * 4 + q] = s; redQ[tok * 4 + q] = sq;
  __syncthreads();
  if (q == 0){
    float ss = redS[tok * 4] + redS[tok * 4 + 1] + redS[tok * 4 + 2] + redS[tok * 4 + 3];
    float qq = redQ[tok * 4] + redQ[tok * 4 + 1] + redQ[tok * 4 + 2] + redQ[tok * 4 + 3];
    float mu = ss * (1.f / 128.f);
    float var = qq * (1.f / 128.f) - mu * mu;
    muS[tok] = mu; rsS[tok] = rsqrtf(var + 1e-5f);
  }
  __syncthreads();
  for (int idx = tid; idx < 64 * 128; idx += 256){
    int c = idx >> 6, t = idx & 63;
    float v = (Cls[t * 129 + c] - muS[t]) * rsS[t] * ln_g[c] + ln_b[c];
    fT[(long)c * NTOK + t0 + t] = v;
  }
}

// ---------------- Kernel 9: InstanceNorm over (H,W) + LeakyReLU -> fnT [128][NTOK]
__global__ void k_inorm(const float* __restrict__ fT, const float* __restrict__ g,
                        const float* __restrict__ bi, float* __restrict__ fnT){
  __shared__ float rs_[256], rq_[256];
  __shared__ float mV, rV;
  int c = blockIdx.x >> 3, b = blockIdx.x & 7;
  int tid = threadIdx.x;
  const float* src = fT + (long)c * NTOK + b * 4096;
  float s = 0.f, q = 0.f;
  for (int i = tid; i < 4096; i += 256){ float v = src[i]; s += v; q += v * v; }
  rs_[tid] = s; rq_[tid] = q;
  __syncthreads();
  for (int st = 128; st > 0; st >>= 1){
    if (tid < st){ rs_[tid] += rs_[tid + st]; rq_[tid] += rq_[tid + st]; }
    __syncthreads();
  }
  if (tid == 0){
    float mu = rs_[0] * (1.f / 4096.f);
    float var = rq_[0] * (1.f / 4096.f) - mu * mu;
    mV = mu; rV = rsqrtf(var + 1e-5f);
  }
  __syncthreads();
  float gg = g[c], bb = bi[c];
  float* dst = fnT + (long)c * NTOK + b * 4096;
  for (int i = tid; i < 4096; i += 256){
    float v = (src[i] - mV) * rV * gg + bb;
    dst[i] = (v >= 0.f) ? v : 0.01f * v;
  }
}

// ---------------- Kernel 10: 3x3 SAME conv as implicit GEMM (K = 9*128), WMMA
__global__ void k_conv2d(const float* __restrict__ fnT, const float* __restrict__ wc,
                         const float* __restrict__ bc, float* __restrict__ out){
  __shared__ __align__(16) bf16_t As[2048];
  __shared__ __align__(16) bf16_t Ws[2048];
  int tid = threadIdx.x, lane = tid & 31, w = tid >> 5;
  int b = blockIdx.x >> 6, y = blockIdx.x & 63;
  int gn0 = blockIdx.y * 64;
  int mt = w >> 1, nb = (w & 1) * 2;
  v8f acc[2] = {v8f_zero(), v8f_zero()};
  for (int kc = 0; kc < 36; ++kc){
    int p = kc >> 2, ci0 = (kc & 3) * 32;
    int dy = p / 3 - 1, dx = p % 3 - 1;
    int yy = y + dy;
    __syncthreads();
    {  // A tile: 64 pixels x 32 in-channels, zero pad at borders, one b128 store per thread
      int px = tid & 63, cb_ = (tid >> 6) << 3;
      int xx = px + dx;
      bool ok = (yy >= 0) && (yy < 64) && (xx >= 0) && (xx < 64);
      const float* pg = fnT + (long)(ci0 + cb_) * NTOK + b * 4096 + yy * 64 + xx;
      float v[8];
#pragma unroll
      for (int i = 0; i < 8; ++i) v[i] = ok ? pg[(long)i * NTOK] : 0.f;
      *(v8bf*)(As + px * 32 + cb_) = pack8(v);
    }
    {  // W tile: 64 out-channels x 32 in-channels at kernel tap p
      int kb = (tid & 3) << 3, n = tid >> 2;
      float v[8];
#pragma unroll
      for (int i = 0; i < 8; ++i)
        v[i] = wc[((long)(gn0 + n) * 128 + ci0 + kb + i) * 9 + p];
      *(v8bf*)(Ws + n * 32 + kb) = pack8(v);
    }
    __syncthreads();
    v16bf af = frag_a(As, mt * 16, lane);
#pragma unroll
    for (int i = 0; i < 2; ++i)
      acc[i] = wmma_bf16(af, frag_b(Ws, (nb + i) * 16, lane), acc[i]);
  }
  int h = lane >> 4, cl = lane & 15;
#pragma unroll
  for (int i = 0; i < 2; ++i){
    int col = gn0 + (nb + i) * 16 + cl;
    float bias = bc[col];
    long base = (long)b * 524288 + (long)col * 4096 + y * 64 + mt * 16 + 8 * h;
#pragma unroll
    for (int r = 0; r < 8; ++r) out[base + r] = acc[i][r] + bias;
  }
}

extern "C" void kernel_launch(void* const* d_in, const int* in_sizes, int n_in,
                              void* d_out, int out_size, void* d_ws, size_t ws_size,
                              hipStream_t stream) {
  (void)in_sizes; (void)n_in; (void)out_size; (void)ws_size;
  const float* x     = (const float*)d_in[0];
  const float* W1    = (const float*)d_in[1];   // proj_in_w (128,64)
  const float* b1    = (const float*)d_in[2];
  const float* W2    = (const float*)d_in[3];   // in_proj_w (512,128)
  const float* cw    = (const float*)d_in[4];   // conv1d_w (256,4)
  const float* cb    = (const float*)d_in[5];
  const float* Wx    = (const float*)d_in[6];   // x_proj_w (136,256)
  const float* Wdt   = (const float*)d_in[7];   // dt_proj_w (256,8)
  const float* bdt   = (const float*)d_in[8];
  const float* A_log = (const float*)d_in[9];
  const float* Dp    = (const float*)d_in[10];
  const float* Wo    = (const float*)d_in[11];  // out_proj_w (128,256)
  const float* Wp    = (const float*)d_in[12];  // proj_out_w (128,128)
  const float* bp    = (const float*)d_in[13];
  const float* lng   = (const float*)d_in[14];
  const float* lnb   = (const float*)d_in[15];
  const float* ing   = (const float*)d_in[16];
  const float* inb   = (const float*)d_in[17];
  const float* wc    = (const float*)d_in[18];  // conv2d_w (128,128,3,3)
  const float* bcv   = (const float*)d_in[19];
  float* outp = (float*)d_out;

  char* ws = (char*)d_ws;
  size_t off = 0;
  bf16_t* t1bf  = (bf16_t*)(ws + off); off += (size_t)NTOK * 128 * sizeof(bf16_t);
  float* xmrT   = (float*)(ws + off);  off += (size_t)256 * NTOK * sizeof(float);
  float* zT     = (float*)(ws + off);  off += (size_t)256 * NTOK * sizeof(float);
  float* xmaT   = (float*)(ws + off);  off += (size_t)256 * NTOK * sizeof(float);
  float* dtlowT = (float*)(ws + off);  off += (size_t)8   * NTOK * sizeof(float);
  float* Bc     = (float*)(ws + off);  off += (size_t)NTOK * 64 * sizeof(float);
  float* Cc     = (float*)(ws + off);  off += (size_t)NTOK * 64 * sizeof(float);
  float* dtT    = (float*)(ws + off);  off += (size_t)256 * NTOK * sizeof(float);
  float* yT     = (float*)(ws + off);  off += (size_t)256 * NTOK * sizeof(float);
  float* o1T    = (float*)(ws + off);  off += (size_t)128 * NTOK * sizeof(float);
  float* fT     = (float*)(ws + off);  off += (size_t)128 * NTOK * sizeof(float);
  float* fnT    = (float*)(ws + off);  off += (size_t)128 * NTOK * sizeof(float);

  dim3 blk(256);
  k_proj_in    <<<dim3(512, 2), blk, 0, stream>>>(x, W1, b1, t1bf);
  k_in_proj    <<<dim3(512, 8), blk, 0, stream>>>(t1bf, W2, xmrT, zT);
  k_conv1d     <<<dim3(2048),   blk, 0, stream>>>(xmrT, cw, cb, xmaT);
  k_x_proj     <<<dim3(512, 3), blk, 0, stream>>>(xmaT, Wx, dtlowT, Bc, Cc);
  k_dt_proj    <<<dim3(512, 4), blk, 0, stream>>>(dtlowT, Wdt, bdt, dtT);
  k_scan       <<<dim3(256),    blk, 0, stream>>>(dtT, xmaT, Bc, Cc, A_log, yT);
  k_out_proj   <<<dim3(512, 2), blk, 0, stream>>>(yT, xmaT, zT, Dp, Wo, o1T);
  k_proj_out_ln<<<dim3(512),    blk, 0, stream>>>(o1T, Wp, bp, lng, lnb, fT);
  k_inorm      <<<dim3(1024),   blk, 0, stream>>>(fT, ing, inb, fnT);
  k_conv2d     <<<dim3(512, 2), blk, 0, stream>>>(fnT, wc, bcv, outp);
}
